// MultiEdgeClassifier_20899310862953
// MI455X (gfx1250) — matrix operand order repaired and checked by reading, not compiled, not run on hardware
//
#include <hip/hip_runtime.h>
#include <hip/hip_bf16.h>

#define HID 128
#define BN_EPS 1e-5f

typedef __attribute__((ext_vector_type(16))) _Float16 v16h;
typedef __attribute__((ext_vector_type(8)))  float    v8f;

// ---------------- graph normalization ----------------
__global__ void k_deg_init(float* deg, int N) {
    int i = blockIdx.x * 256 + threadIdx.x;
    if (i < N) deg[i] = 1.0f;               // self-loop
}
__global__ void k_deg_count(const int* __restrict__ dst, float* deg, int E) {
    int e = blockIdx.x * 256 + threadIdx.x;
    if (e < E) atomicAdd(&deg[dst[e]], 1.0f);
}
__global__ void k_dinv(float* deg, int N) { // in place: deg >= 1 always
    int i = blockIdx.x * 256 + threadIdx.x;
    if (i < N) deg[i] = rsqrtf(deg[i]);
}
__global__ void k_norm(const int* __restrict__ src, const int* __restrict__ dst,
                       const float* __restrict__ dinv, float* __restrict__ norm, int E) {
    int e = blockIdx.x * 256 + threadIdx.x;
    if (e < E) norm[e] = dinv[src[e]] * dinv[dst[e]];
}

// ---------------- input embedding: xe = x @ embW + embB ----------------
__global__ void k_embed(const float* __restrict__ x, const float* __restrict__ W,
                        const float* __restrict__ b, float* __restrict__ xe, int N) {
    __shared__ float sW[16 * HID];
    __shared__ float sb[HID];
    int tid = threadIdx.x;
    for (int i = tid; i < 16 * HID; i += 256) sW[i] = W[i];
    if (tid < HID) sb[tid] = b[tid];
    __syncthreads();
    int c = tid & (HID - 1);
    int sub = tid >> 7; // 2 nodes per block iteration
    for (int n = blockIdx.x * 2 + sub; n < N; n += gridDim.x * 2) {
        const float* xr = x + (size_t)n * 16;
        float acc = sb[c];
#pragma unroll
        for (int k = 0; k < 16; ++k) acc += xr[k] * sW[k * HID + c];
        xe[(size_t)n * HID + c] = acc;
    }
}

// ---------------- layer-weight softmax ----------------
__global__ void k_softmax(const float* __restrict__ lw, float* __restrict__ w, int L) {
    if (threadIdx.x == 0 && blockIdx.x == 0) {
        float m = -1e30f;
        for (int i = 0; i < L; ++i) m = fmaxf(m, lw[i]);
        float s = 0.f;
        for (int i = 0; i < L; ++i) s += expf(lw[i] - m);
        for (int i = 0; i < L; ++i) w[i] = expf(lw[i] - m) / s;
    }
}

// ---------------- pack conv_W (f32 [L][K=128][N=128]) into WMMA f16 B layout --------
// Per layer blob (16384 halves): index = ((nt*4+kc)*32 + lane)*16 + j
//   lane<16:  col = nt*16 + lane,     K = kc*32 + j        (j=0..15)
//   lane>=16: col = nt*16 + lane-16,  K = kc*32 + 16 + j
__global__ void k_packW(const float* __restrict__ W, _Float16* __restrict__ Bp, int L) {
    int idx = blockIdx.x * 256 + threadIdx.x;
    int total = L * HID * HID;
    if (idx >= total) return;
    int l   = idx / (HID * HID);
    int rem = idx - l * HID * HID;
    int k   = rem / HID;
    int n   = rem - k * HID;
    int nt = n >> 4, kc = k >> 5;
    int kk = k & 31;
    int lane = (n & 15) + (kk & 16);
    int j = kk & 15;
    size_t off = (size_t)l * (HID * HID) + (size_t)(((nt * 4 + kc) * 32 + lane) * 16 + j);
    Bp[off] = (_Float16)W[idx];
}

// ---------------- GEMM: h = xe(f32, cvt->f16) @ Bpack_layer, f32 accum via WMMA -----
__global__ void k_gemm(const float* __restrict__ xe, const _Float16* __restrict__ Bp,
                       float* __restrict__ h, int mtiles) {
    __shared__ alignas(32) uint4 ldsB[2048]; // 32KB: entire packed 128x128 f16 weight
    int tid = threadIdx.x;
    const uint4* gB = (const uint4*)Bp;
#pragma unroll
    for (int i = 0; i < 8; ++i) ldsB[tid + i * 256] = gB[tid + i * 256];
    __syncthreads();

    int lane = tid & 31;
    int wid  = tid >> 5;
    int halfsel = (lane & 16) ? 1 : 0;   // lane>=16 holds K-offset +16 half of A
    int mrow = lane & 15;

    for (int mt = blockIdx.x * 8 + wid; mt < mtiles; mt += gridDim.x * 8) {
        int row = mt * 16 + mrow;
        const float* xr = xe + (size_t)row * HID + (halfsel ? 16 : 0);
        v16h a[4];
#pragma unroll
        for (int kc = 0; kc < 4; ++kc) {
            const float* p = xr + kc * 32;
            float4 f0 = *(const float4*)(p + 0);
            float4 f1 = *(const float4*)(p + 4);
            float4 f2 = *(const float4*)(p + 8);
            float4 f3 = *(const float4*)(p + 12);
            v16h t;
            t[0]  = (_Float16)f0.x; t[1]  = (_Float16)f0.y;
            t[2]  = (_Float16)f0.z; t[3]  = (_Float16)f0.w;
            t[4]  = (_Float16)f1.x; t[5]  = (_Float16)f1.y;
            t[6]  = (_Float16)f1.z; t[7]  = (_Float16)f1.w;
            t[8]  = (_Float16)f2.x; t[9]  = (_Float16)f2.y;
            t[10] = (_Float16)f2.z; t[11] = (_Float16)f2.w;
            t[12] = (_Float16)f3.x; t[13] = (_Float16)f3.y;
            t[14] = (_Float16)f3.z; t[15] = (_Float16)f3.w;
            a[kc] = t;
        }

        // One base pointer; all stores below are immediate offsets off it.
        float* hp0 = h + (size_t)(mt * 16 + (halfsel ? 8 : 0)) * HID + mrow;

        // Keep this loop rolled: B must stream from LDS per iteration, not be
        // hoisted/spilled to scratch (32 live v16h tiles would blow VGPRs).
#pragma unroll 1
        for (int nt = 0; nt < 8; ++nt) {
            v8f acc = {};
#pragma unroll
            for (int kc = 0; kc < 4; ++kc) {
                const v16h* bp =
                    reinterpret_cast<const v16h*>(&ldsB[((nt * 4 + kc) * 32 + lane) * 2]);
                acc = __builtin_amdgcn_wmma_f32_16x16x32_f16(
                    false, a[kc], false, *bp, (short)0, acc, false, false);
            }
            float* hp = hp0 + nt * 16;
#pragma unroll
            for (int r = 0; r < 8; ++r) {
                hp[r * HID] = acc[r];
            }
        }
    }
}

// Naive fallback for M tail rows (only launched when N % 16 != 0)
__global__ void k_gemm_tail(const float* __restrict__ xe, const float* __restrict__ W,
                            float* __restrict__ h, int row0, int N) {
    int idx = blockIdx.x * 256 + threadIdx.x;
    int n = row0 + idx / HID;
    int c = idx % HID;
    if (n >= N) return;
    const float* xr = xe + (size_t)n * HID;
    float acc = 0.f;
    for (int k = 0; k < HID; ++k) acc += xr[k] * W[k * HID + c];
    h[(size_t)n * HID + c] = acc;
}

// ---------------- agg init: bias + self-loop contribution ----------------
__global__ void k_agg_init(const float* __restrict__ h, const float* __restrict__ dinv,
                           const float* __restrict__ cb, float* __restrict__ agg, int N) {
    int idx = blockIdx.x * 256 + threadIdx.x;
    if (idx >= N * HID) return;
    int n = idx >> 7;
    int c = idx & (HID - 1);
    float di = dinv[n];
    agg[idx] = cb[c] + h[idx] * di * di;
}

// ---------------- edge scatter: agg[dst] += h[src] * norm (wave per edge) ----------
__global__ void k_scatter(const int* __restrict__ esrc, const int* __restrict__ edst,
                          const float* __restrict__ norm, const float* __restrict__ h,
                          float* __restrict__ agg, int E) {
    int t = blockIdx.x * 256 + threadIdx.x;
    int e = t >> 5;
    if (e >= E) return;
    int lane = t & 31;
    int s = esrc[e], d = edst[e];
    float nv = norm[e];
    float4 v = *(const float4*)(h + (size_t)s * HID + lane * 4);
    float* ap = agg + (size_t)d * HID + lane * 4;
    atomicAdd(ap + 0, v.x * nv);
    atomicAdd(ap + 1, v.y * nv);
    atomicAdd(ap + 2, v.z * nv);
    atomicAdd(ap + 3, v.w * nv);
}

// ---------------- batchnorm ----------------
__global__ void k_zero(float* p, int n) {
    int i = blockIdx.x * 256 + threadIdx.x;
    if (i < n) p[i] = 0.f;
}
__global__ void k_bn_reduce(const float* __restrict__ agg, float* __restrict__ sum,
                            float* __restrict__ sumsq, int N) {
    int gid = blockIdx.x * blockDim.x + threadIdx.x;
    int c = gid & (HID - 1);
    int r = gid >> 7;
    int stride = (gridDim.x * blockDim.x) >> 7;
    float s = 0.f, q = 0.f;
    for (; r < N; r += stride) {
        float v = agg[(size_t)r * HID + c];
        s += v;
        q += v * v;
    }
    atomicAdd(&sum[c], s);
    atomicAdd(&sumsq[c], q);
}
__global__ void k_bn_fin(const float* __restrict__ sum, const float* __restrict__ sumsq,
                         const float* __restrict__ gamma, const float* __restrict__ beta,
                         float* __restrict__ scale, float* __restrict__ shift, int N) {
    int c = threadIdx.x;
    if (c >= HID) return;
    float inv_n = 1.0f / (float)N;
    float mean = sum[c] * inv_n;
    float var = fmaxf(sumsq[c] * inv_n - mean * mean, 0.f);
    float inv = rsqrtf(var + BN_EPS);
    float sc = gamma[c] * inv;
    scale[c] = sc;
    shift[c] = beta[c] - mean * sc;
}

// ---------------- fused BN-apply + ReLU + weighted residual ----------------
__global__ void k_update(float* __restrict__ xe, const float* __restrict__ agg,
                         const float* __restrict__ scale, const float* __restrict__ shift,
                         const float* __restrict__ wsm, int layer, int total) {
    int idx = blockIdx.x * 256 + threadIdx.x;
    if (idx >= total) return;
    int c = idx & (HID - 1);
    float v = agg[idx] * scale[c] + shift[c];
    v = fmaxf(v, 0.f);
    xe[idx] += wsm[layer] * v;
}

// ---------------- final edge classifier (wave per output edge) ----------------
__global__ void k_fc(const float* __restrict__ xe, const int* __restrict__ es,
                     const int* __restrict__ ed, const float* __restrict__ fcW,
                     const float* __restrict__ fcb, float* __restrict__ out, int EO) {
    int t = blockIdx.x * 256 + threadIdx.x;
    int e = t >> 5;
    if (e >= EO) return;
    int lane = t & 31;
    int s = es[e], d = ed[e];
    const float* xs = xe + (size_t)s * HID;
    const float* xd = xe + (size_t)d * HID;
    float a0 = 0.f, a1 = 0.f;
#pragma unroll
    for (int c = lane; c < HID; c += 32) {
        float v0 = xs[c], v1 = xd[c];
        a0 += v0 * fcW[c * 2 + 0] + v1 * fcW[(HID + c) * 2 + 0];
        a1 += v0 * fcW[c * 2 + 1] + v1 * fcW[(HID + c) * 2 + 1];
    }
#pragma unroll
    for (int off = 16; off; off >>= 1) {
        a0 += __shfl_down(a0, off, 32);
        a1 += __shfl_down(a1, off, 32);
    }
    if (lane == 0) {
        out[(size_t)e * 2 + 0] = a0 + fcb[0];
        out[(size_t)e * 2 + 1] = a1 + fcb[1];
    }
}

extern "C" void kernel_launch(void* const* d_in, const int* in_sizes, int n_in,
                              void* d_out, int out_size, void* d_ws, size_t ws_size,
                              hipStream_t stream) {
    (void)n_in; (void)out_size; (void)ws_size;
    const float* x     = (const float*)d_in[0];
    const int*   ei    = (const int*)d_in[1];
    const int*   eo    = (const int*)d_in[2];
    const float* embW  = (const float*)d_in[3];
    const float* embB  = (const float*)d_in[4];
    const float* convW = (const float*)d_in[5];
    const float* convB = (const float*)d_in[6];
    const float* gamma = (const float*)d_in[7];
    const float* beta  = (const float*)d_in[8];
    const float* lw    = (const float*)d_in[9];
    const float* fcW   = (const float*)d_in[10];
    const float* fcb   = (const float*)d_in[11];
    float* out = (float*)d_out;

    const int N  = in_sizes[0] / 16;
    const int E  = in_sizes[1] / 2;
    const int EO = in_sizes[2] / 2;
    const int L  = in_sizes[9];

    char* base = (char*)d_ws;
    size_t off = 0;
    auto carve = [&](size_t bytes) -> char* {
        char* p = base + off;
        off = (off + bytes + 255) & ~(size_t)255;
        return p;
    };
    float*    dinv    = (float*)carve((size_t)N * 4);
    float*    norm    = (float*)carve((size_t)E * 4);
    float*    xe      = (float*)carve((size_t)N * HID * 4);
    float*    hbuf    = (float*)carve((size_t)N * HID * 4);
    float*    agg     = (float*)carve((size_t)N * HID * 4);
    _Float16* Bp      = (_Float16*)carve((size_t)L * HID * HID * 2);
    float*    wsm     = (float*)carve(64);
    float*    bnstats = (float*)carve(1024);     // sum[128] | sumsq[128]
    float*    scale   = (float*)carve(512);
    float*    shift   = (float*)carve(512);

    const int* esrc = ei;
    const int* edst = ei + E;

    k_deg_init<<<(N + 255) / 256, 256, 0, stream>>>(dinv, N);
    k_deg_count<<<(E + 255) / 256, 256, 0, stream>>>(edst, dinv, E);
    k_dinv<<<(N + 255) / 256, 256, 0, stream>>>(dinv, N);
    k_norm<<<(E + 255) / 256, 256, 0, stream>>>(esrc, edst, dinv, norm, E);
    k_embed<<<2048, 256, 0, stream>>>(x, embW, embB, xe, N);
    k_softmax<<<1, 64, 0, stream>>>(lw, wsm, L);
    k_packW<<<(L * HID * HID + 255) / 256, 256, 0, stream>>>(convW, Bp, L);

    int mtiles = N / 16;
    int tail = N - mtiles * 16;
    int total = N * HID;

    for (int i = 0; i < L; ++i) {
        k_gemm<<<(mtiles + 7) / 8, 256, 0, stream>>>(
            xe, Bp + (size_t)i * HID * HID, hbuf, mtiles);
        if (tail)
            k_gemm_tail<<<(tail * HID + 255) / 256, 256, 0, stream>>>(
                xe, convW + (size_t)i * HID * HID, hbuf, mtiles * 16, N);
        k_agg_init<<<(total + 255) / 256, 256, 0, stream>>>(
            hbuf, dinv, convB + (size_t)i * HID, agg, N);
        k_scatter<<<(E + 7) / 8, 256, 0, stream>>>(esrc, edst, norm, hbuf, agg, E);
        k_zero<<<1, 256, 0, stream>>>(bnstats, 256);
        k_bn_reduce<<<1024, 256, 0, stream>>>(agg, bnstats, bnstats + HID, N);
        k_bn_fin<<<1, 128, 0, stream>>>(bnstats, bnstats + HID,
                                        gamma + (size_t)i * HID, beta + (size_t)i * HID,
                                        scale, shift, N);
        k_update<<<(total + 255) / 256, 256, 0, stream>>>(
            xe, agg, scale, shift, wsm, i, total);
    }

    k_fc<<<(EO * 32 + 255) / 256, 256, 0, stream>>>(xe, eo, eo + EO, fcW, fcb, out, EO);
}
